// DAGERC_v2_44727789421061
// MI455X (gfx1250) — compile-verified, matrix operand order checked by reading, not run on hardware
//
#include <hip/hip_runtime.h>
#include <hip/hip_bf16.h>
#include <math.h>

// ---------------- problem dims ----------------
#define BT   48
#define NN   96
#define EE   1024
#define DD   300
#define DP   320     // D padded to mult of 32 (WMMA K)
#define GG   900     // 3*D gate width
#define GP   912     // padded to mult of 16 (WMMA N)
#define LL   4
#define NC   7
#define BN   (BT*NN)          // 4608 rows
#define CATR 3724             // concat width real
#define CATP 3744             // concat width padded to mult 32

typedef __attribute__((ext_vector_type(16))) _Float16 v16h;
typedef __attribute__((ext_vector_type(8)))  _Float16 v8h;
typedef __attribute__((ext_vector_type(8)))  float    v8f;

union Frag16 { v16h v; v8h h[2]; };

// ---------------- WMMA helpers ----------------
// 16-bit A/B fragment layout (CDNA5 ISA 7.12.2): lane%16 = row (A) / col (B),
// per-lane K chunks at k0=(lane/16)*8 and k0+16, 8 contiguous halves each.
__device__ __forceinline__ v8f wmma_tile_f16(const _Float16* __restrict__ A, int lda,
                                             const _Float16* __restrict__ B, int ldb,
                                             int kSteps, int m0, int n0, int lane) {
  const int lh = lane & 15;
  const int kb = (lane >> 4) << 3;
  v8f acc = {};
  const _Float16* arow = A + (size_t)(m0 + lh) * lda;
  const _Float16* brow = B + (size_t)(n0 + lh) * ldb;
  for (int kk = 0; kk < kSteps; ++kk) {
    const int k0 = (kk << 5) + kb;
    Frag16 a, b;
    a.h[0] = *(const v8h*)(arow + k0);
    a.h[1] = *(const v8h*)(arow + k0 + 16);
    b.h[0] = *(const v8h*)(brow + k0);
    b.h[1] = *(const v8h*)(brow + k0 + 16);
    if (kk + 1 < kSteps) __builtin_prefetch(brow + k0 + 32, 0, 1); // global_prefetch_b8
    acc = __builtin_amdgcn_wmma_f32_16x16x32_f16(false, a.v, false, b.v,
                                                 (short)0, acc, false, false);
  }
  return acc;
}

// D-tile layout: lane%16 = col, rows = (lane/16)*8 + r for r in 0..7
__device__ __forceinline__ void store_tile(float* __restrict__ C, int ldc,
                                           int m0, int n0, v8f acc,
                                           const float* __restrict__ bias,
                                           int nReal, bool relu, int lane) {
  const int lh = lane & 15;
  const int mb = (lane >> 4) << 3;
  const int n  = n0 + lh;
  if (n < nReal) {
    const float bv = bias ? bias[n] : 0.f;
#pragma unroll
    for (int r = 0; r < 8; ++r) {
      float v = acc[r] + bv;
      if (relu) v = fmaxf(v, 0.f);
      C[(size_t)(m0 + mb + r) * ldc + n] = v;
    }
  }
}

__device__ __forceinline__ float wave_sum(float s) {
#pragma unroll
  for (int o = 16; o > 0; o >>= 1) s += __shfl_xor(s, o, 32);
  return s;
}
__device__ __forceinline__ float wave_max(float s) {
#pragma unroll
  for (int o = 16; o > 0; o >>= 1) s = fmaxf(s, __shfl_xor(s, o, 32));
  return s;
}

// ---------------- conversion / layout kernels ----------------
__global__ void k_cvt_f16(const float* __restrict__ s, _Float16* __restrict__ d, int n) {
  int t = blockIdx.x * blockDim.x + threadIdx.x;
  if (t < n) d[t] = (_Float16)s[t];
}

// src (R,C) f32 row-major -> dst (Rp,Cp) f16 zero-padded
__global__ void k_pad_cvt(const float* __restrict__ s, _Float16* __restrict__ d,
                          int R, int C, int Rp, int Cp) {
  int t = blockIdx.x * blockDim.x + threadIdx.x;
  if (t >= Rp * Cp) return;
  int r = t / Cp, c = t % Cp;
  d[t] = (r < R && c < C) ? (_Float16)s[(size_t)r * C + c] : (_Float16)0.f;
}

// src (K,N) f32 -> dst (Np,Kp) f16 : dst[n*Kp+k] = src[k*N+n]  (N-major, K-contig)
__global__ void k_transpose_cvt(const float* __restrict__ s, _Float16* __restrict__ d,
                                int K, int N, int Np, int Kp) {
  int t = blockIdx.x * blockDim.x + threadIdx.x;
  if (t >= Np * Kp) return;
  int n = t / Kp, k = t % Kp;
  d[t] = (n < N && k < K) ? (_Float16)s[(size_t)k * N + n] : (_Float16)0.f;
}

// ---------------- generic WMMA GEMM:  C = act(A(M,Kp) * B(Np,Kp)^T + bias) ----------------
__global__ void k_gemm_f16(const _Float16* __restrict__ A, const _Float16* __restrict__ B,
                           const float* __restrict__ bias, float* __restrict__ C,
                           int M, int Kp, int Np, int Nreal, int ldc, int relu) {
  const int wid  = (blockIdx.x * blockDim.x + threadIdx.x) >> 5;
  const int lane = threadIdx.x & 31;
  const int mt = M >> 4, nt = Np >> 4;
  if (wid >= mt * nt) return;
  const int m0 = (wid / nt) << 4;
  const int n0 = (wid % nt) << 4;
  v8f acc = wmma_tile_f16(A, Kp, B, Kp, Kp >> 5, m0, n0, lane);
  store_tile(C, ldc, m0, n0, acc, bias, Nreal, relu != 0, lane);
}

// ---------------- per-layer recurrent scan (single 1024-thread block) ----------------
__device__ __forceinline__ void mm_lds(const _Float16* __restrict__ As,
                                       const _Float16* __restrict__ W,
                                       const float* __restrict__ bias,
                                       float* __restrict__ G, int wid, int lane) {
  for (int t = wid; t < 3 * 57; t += 32) {          // 48x912 output, 171 tiles
    const int m0 = (t / 57) << 4;
    const int n0 = (t % 57) << 4;
    v8f acc = wmma_tile_f16(As, DP, W, DP, DP >> 5, m0, n0, lane);
    store_tile(G, GP, m0, n0, acc, bias, GG, false, lane);
  }
}

__global__ void __launch_bounds__(1024, 1)
k_layer_scan(const float* __restrict__ Cin, const float* __restrict__ Hin,
             const float* __restrict__ adj,
             const _Float16* __restrict__ cwi, const _Float16* __restrict__ cwh,
             const _Float16* __restrict__ pwi, const _Float16* __restrict__ pwh,
             const float* __restrict__ cbi, const float* __restrict__ cbh,
             const float* __restrict__ pbi, const float* __restrict__ pbh,
             const float* __restrict__ wq, const float* __restrict__ wk,
             const float* __restrict__ gbp,
             float* __restrict__ CL, float* __restrict__ Pout,
             float* __restrict__ Mbuf, float* __restrict__ g0, float* __restrict__ g1,
             float* __restrict__ g2, float* __restrict__ g3) {
  __shared__ __align__(16) _Float16 As[BT * DP];   // 30 KB, aliased with aw below
  __shared__ float pk[BT * NN];                    // 18 KB, persistent key projections
  __shared__ float qws[BT];
  float* aw = (float*)As;                          // attention weights 48x96 (phase A/B only)

  const int tid  = threadIdx.x;
  const int lane = tid & 31;
  const int wid  = tid >> 5;
  const float gb = gbp[0];

  for (int i = 0; i < NN; ++i) {
    // ---- Phase A/B: attention + context M ----
    if (i > 0) {
      for (int b = wid; b < BT; b += 32) {         // qw[b] = Cin[b,i,:] . wq
        const float* q = Cin + ((size_t)b * NN + i) * DD;
        float s = 0.f;
        for (int d = lane; d < DD; d += 32) s += q[d] * wq[d];
        s = wave_sum(s);
        if (lane == 0) qws[b] = s;
      }
      __syncthreads();
      for (int b = wid; b < BT; b += 32) {         // masked softmax over j<i
        const float* arow = adj + ((size_t)b * NN + i) * NN;
        const int j0 = lane, j1 = lane + 32, j2 = lane + 64;
        const bool m0 = (j0 < i) && (arow[j0] != 0.f);
        const bool m1 = (j1 < i) && (arow[j1] != 0.f);
        const bool m2 = (j2 < i) && (arow[j2] != 0.f);
        const float qb = qws[b];
        float a0 = m0 ? qb + pk[b * NN + j0] + gb : -3.4e38f;
        float a1 = m1 ? qb + pk[b * NN + j1] + gb : -3.4e38f;
        float a2 = m2 ? qb + pk[b * NN + j2] + gb : -3.4e38f;
        float mx = wave_max(fmaxf(a0, fmaxf(a1, a2)));
        float e0 = m0 ? __expf(a0 - mx) : 0.f;
        float e1 = m1 ? __expf(a1 - mx) : 0.f;
        float e2 = m2 ? __expf(a2 - mx) : 0.f;
        float inv = 1.f / wave_sum(e0 + e1 + e2);
        aw[b * NN + j0] = e0 * inv;
        aw[b * NN + j1] = e1 * inv;
        aw[b * NN + j2] = e2 * inv;
      }
      __syncthreads();
      for (int t = tid; t < BT * DD; t += 1024) {  // M[b,d] = sum_j w[b,j]*P[b,j,d]
        const int b = t / DD, d = t % DD;
        const float* Pb = Pout + (size_t)b * NN * DD + d;
        const float* wb = aw + b * NN;
        float s = 0.f;
        for (int j = 0; j < i; ++j) s += wb[j] * Pb[(size_t)j * DD];
        Mbuf[t] = s;
      }
      __syncthreads();
    } else {
      for (int t = tid; t < BT * DD; t += 1024) Mbuf[t] = 0.f;
      __syncthreads();
    }

    // ---- Phase C1: g0 = q @ cwi^T + cbi ----
    for (int t = tid; t < BT * DP; t += 1024) {
      const int b = t / DP, k = t % DP;
      As[t] = (k < DD) ? (_Float16)Cin[((size_t)b * NN + i) * DD + k] : (_Float16)0.f;
    }
    __syncthreads();
    mm_lds(As, cwi, cbi, g0, wid, lane);
    __syncthreads();

    // ---- Phase C2: stage M; g1 = M @ cwh^T + cbh ; g2 = M @ pwi^T + pbi ----
    for (int t = tid; t < BT * DP; t += 1024) {
      const int b = t / DP, k = t % DP;
      As[t] = (k < DD) ? (_Float16)Mbuf[b * DD + k] : (_Float16)0.f;
    }
    __syncthreads();
    mm_lds(As, cwh, cbh, g1, wid, lane);
    mm_lds(As, pwi, pbi, g2, wid, lane);
    __syncthreads();

    // ---- Gate C: c_i -> CL[:,i] ----
    for (int t = tid; t < BT * DD; t += 1024) {
      const int b = t / DD, d = t % DD;
      const float* G0 = g0 + (size_t)b * GP;
      const float* G1 = g1 + (size_t)b * GP;
      const float r = 1.f / (1.f + __expf(-(G0[d] + G1[d])));
      const float z = 1.f / (1.f + __expf(-(G0[DD + d] + G1[DD + d])));
      const float n = tanhf(G0[2 * DD + d] + r * G1[2 * DD + d]);
      CL[((size_t)b * NN + i) * DD + d] = (1.f - z) * n + z * Mbuf[t];
    }

    // ---- Phase C3: g3 = h_p @ pwh^T + pbh  (h_p = Hin[:,i], or Cin[:,0] at i==0) ----
    const float* hp_src = (i == 0) ? Cin : Hin;
    __syncthreads();
    for (int t = tid; t < BT * DP; t += 1024) {
      const int b = t / DP, k = t % DP;
      As[t] = (k < DD) ? (_Float16)hp_src[((size_t)b * NN + i) * DD + k] : (_Float16)0.f;
    }
    __syncthreads();
    mm_lds(As, pwh, pbh, g3, wid, lane);
    __syncthreads();

    // ---- Gate P: p_i -> Pout[:,i] ----
    for (int t = tid; t < BT * DD; t += 1024) {
      const int b = t / DD, d = t % DD;
      const float* G2 = g2 + (size_t)b * GP;
      const float* G3 = g3 + (size_t)b * GP;
      const float r = 1.f / (1.f + __expf(-(G2[d] + G3[d])));
      const float z = 1.f / (1.f + __expf(-(G2[DD + d] + G3[DD + d])));
      const float n = tanhf(G2[2 * DD + d] + r * G3[2 * DD + d]);
      const float h = hp_src[((size_t)b * NN + i) * DD + d];
      Pout[((size_t)b * NN + i) * DD + d] = (1.f - z) * n + z * h;
    }
    __syncthreads();
    for (int b = wid; b < BT; b += 32) {           // pk[b,i] = p_i . wk (incremental)
      const float* p = Pout + ((size_t)b * NN + i) * DD;
      float s = 0.f;
      for (int d = lane; d < DD; d += 32) s += p[d] * wk[d];
      s = wave_sum(s);
      if (lane == 0) pk[b * NN + i] = s;
    }
    __syncthreads();
  }
}

// ---------------- concat + head ----------------
__global__ void k_concat_f16(const float* __restrict__ HP, const float* __restrict__ feats,
                             _Float16* __restrict__ out) {
  size_t t = (size_t)blockIdx.x * blockDim.x + threadIdx.x;
  const size_t total = (size_t)BN * CATP;
  if (t >= total) return;
  const int r = (int)(t / CATP), c = (int)(t % CATP);
  float v = 0.f;
  if (c < 9 * DD) {
    const int which = c / DD, d = c % DD;
    v = HP[(size_t)which * BN * DD + (size_t)r * DD + d];
  } else if (c < CATR) {
    v = feats[(size_t)r * EE + (c - 9 * DD)];
  }
  out[t] = (_Float16)v;
}

__global__ void k_mlp2(const float* __restrict__ X2, const float* __restrict__ w2,
                       const float* __restrict__ b2, float* __restrict__ out) {
  int t = blockIdx.x * blockDim.x + threadIdx.x;
  if (t >= BN * NC) return;
  const int r = t / NC, c = t % NC;
  float s = b2[c];
  for (int d = 0; d < DD; ++d) s += X2[(size_t)r * DD + d] * w2[d * NC + c];
  out[t] = s;
}

// ---------------- host side ----------------
static inline int cdiv(long long a, long long b) { return (int)((a + b - 1) / b); }

extern "C" void kernel_launch(void* const* d_in, const int* in_sizes, int n_in,
                              void* d_out, int out_size, void* d_ws, size_t ws_size,
                              hipStream_t stream) {
  (void)in_sizes; (void)n_in; (void)out_size; (void)ws_size;
  const float* features = (const float*)d_in[0];
  const float* adj      = (const float*)d_in[1];
  // d_in[2] s_mask unused by reference
  const float* fc1_w   = (const float*)d_in[3];
  const float* fc1_b   = (const float*)d_in[4];
  const float* gc_wih  = (const float*)d_in[5];
  const float* gc_whh  = (const float*)d_in[6];
  const float* gc_bih  = (const float*)d_in[7];
  const float* gc_bhh  = (const float*)d_in[8];
  const float* gp_wih  = (const float*)d_in[9];
  const float* gp_whh  = (const float*)d_in[10];
  const float* gp_bih  = (const float*)d_in[11];
  const float* gp_bhh  = (const float*)d_in[12];
  const float* gat_wq  = (const float*)d_in[13];
  const float* gat_wk  = (const float*)d_in[14];
  const float* gat_b   = (const float*)d_in[15];
  const float* mlp_w0  = (const float*)d_in[16];
  const float* mlp_b0  = (const float*)d_in[17];
  const float* mlp_w1  = (const float*)d_in[18];
  const float* mlp_b1  = (const float*)d_in[19];
  const float* mlp_w2  = (const float*)d_in[20];
  const float* mlp_b2  = (const float*)d_in[21];
  float* out = (float*)d_out;

  // ---- workspace carve-up ----
  char* p = (char*)d_ws;
  auto alloc = [&](size_t bytes) -> void* {
    void* r = (void*)p;
    p += (bytes + 255) & ~(size_t)255;
    return r;
  };
  _Float16* fA   = (_Float16*)alloc((size_t)BN * EE * 2);        // features f16
  _Float16* w1T  = (_Float16*)alloc((size_t)304 * EE * 2);       // fc1_w^T padded
  float*    HP   = (float*)   alloc((size_t)9 * BN * DD * 4);    // H0,C1,P1,C2,P2,C3,P3,C4,P4
  _Float16* gruW = (_Float16*)alloc((size_t)16 * GP * DP * 2);   // 4 layers x {cwi,cwh,pwi,pwh}
  float*    Mbuf = (float*)   alloc((size_t)BT * DD * 4);
  float*    g0   = (float*)   alloc((size_t)BT * GP * 4);
  float*    g1   = (float*)   alloc((size_t)BT * GP * 4);
  float*    g2   = (float*)   alloc((size_t)BT * GP * 4);
  float*    g3   = (float*)   alloc((size_t)BT * GP * 4);
  _Float16* Hcat = (_Float16*)alloc((size_t)BN * CATP * 2);
  _Float16* w0T  = (_Float16*)alloc((size_t)304 * CATP * 2);
  float*    X1   = (float*)   alloc((size_t)BN * DD * 4);
  _Float16* X1f  = (_Float16*)alloc((size_t)BN * DP * 2);
  _Float16* w1mT = (_Float16*)alloc((size_t)304 * DP * 2);
  float*    X2   = (float*)   alloc((size_t)BN * DD * 4);

  const int TB = 256;

  // ---- precision staging ----
  k_cvt_f16<<<cdiv((long long)BN * EE, TB), TB, 0, stream>>>(features, fA, BN * EE);
  k_transpose_cvt<<<cdiv(304 * EE, TB), TB, 0, stream>>>(fc1_w, w1T, EE, DD, 304, EE);
  for (int l = 0; l < LL; ++l) {
    const size_t wo = (size_t)l * GG * DD, bo = (size_t)l * GG;
    (void)bo;
    k_pad_cvt<<<cdiv(GP * DP, TB), TB, 0, stream>>>(gc_wih + wo, gruW + (size_t)(l * 4 + 0) * GP * DP, GG, DD, GP, DP);
    k_pad_cvt<<<cdiv(GP * DP, TB), TB, 0, stream>>>(gc_whh + wo, gruW + (size_t)(l * 4 + 1) * GP * DP, GG, DD, GP, DP);
    k_pad_cvt<<<cdiv(GP * DP, TB), TB, 0, stream>>>(gp_wih + wo, gruW + (size_t)(l * 4 + 2) * GP * DP, GG, DD, GP, DP);
    k_pad_cvt<<<cdiv(GP * DP, TB), TB, 0, stream>>>(gp_whh + wo, gruW + (size_t)(l * 4 + 3) * GP * DP, GG, DD, GP, DP);
  }

  // ---- H0 = relu(features @ fc1_w + b)  -> HP slot 0 ----
  {
    const int waves = (BN / 16) * (304 / 16);
    k_gemm_f16<<<cdiv((long long)waves * 32, TB), TB, 0, stream>>>(
        fA, w1T, fc1_b, HP, BN, EE, 304, DD, DD, 1);
  }

  // ---- 4 recurrent layers (persistent single-block scans) ----
  const int cinSlot[LL] = {0, 1, 3, 5};  // C_list[l]
  const int hinSlot[LL] = {0, 1, 2, 3};  // H_list[l]
  for (int l = 0; l < LL; ++l) {
    const float* Cin = HP + (size_t)cinSlot[l] * BN * DD;
    const float* Hin = HP + (size_t)hinSlot[l] * BN * DD;
    float* CLp = HP + (size_t)(2 * l + 1) * BN * DD;
    float* Pp  = HP + (size_t)(2 * l + 2) * BN * DD;
    k_layer_scan<<<1, 1024, 0, stream>>>(
        Cin, Hin, adj,
        gruW + (size_t)(l * 4 + 0) * GP * DP, gruW + (size_t)(l * 4 + 1) * GP * DP,
        gruW + (size_t)(l * 4 + 2) * GP * DP, gruW + (size_t)(l * 4 + 3) * GP * DP,
        gc_bih + (size_t)l * GG, gc_bhh + (size_t)l * GG,
        gp_bih + (size_t)l * GG, gp_bhh + (size_t)l * GG,
        gat_wq + (size_t)l * DD, gat_wk + (size_t)l * DD, gat_b + l,
        CLp, Pp, Mbuf, g0, g1, g2, g3);
  }

  // ---- head: concat -> MLP ----
  k_concat_f16<<<cdiv((long long)BN * CATP, TB), TB, 0, stream>>>(HP, features, Hcat);
  k_transpose_cvt<<<cdiv(304 * CATP, TB), TB, 0, stream>>>(mlp_w0, w0T, CATR, DD, 304, CATP);
  {
    const int waves = (BN / 16) * (304 / 16);
    k_gemm_f16<<<cdiv((long long)waves * 32, TB), TB, 0, stream>>>(
        Hcat, w0T, mlp_b0, X1, BN, CATP, 304, DD, DD, 1);
  }
  k_pad_cvt<<<cdiv(BN * DP, TB), TB, 0, stream>>>(X1, X1f, BN, DD, BN, DP);
  k_transpose_cvt<<<cdiv(304 * DP, TB), TB, 0, stream>>>(mlp_w1, w1mT, DD, DD, 304, DP);
  {
    const int waves = (BN / 16) * (304 / 16);
    k_gemm_f16<<<cdiv((long long)waves * 32, TB), TB, 0, stream>>>(
        X1f, w1mT, mlp_b1, X2, BN, DP, 304, DD, DD, 1);
  }
  k_mlp2<<<cdiv(BN * NC, TB), TB, 0, stream>>>(X2, mlp_w2, mlp_b2, out);
}
